// Attention_61753039782136
// MI455X (gfx1250) — matrix operand order repaired and checked by reading, not compiled, not run on hardware
//
#include <hip/hip_runtime.h>
#include <hip/hip_bf16.h>

typedef __attribute__((ext_vector_type(16))) _Float16 v16h;
typedef __attribute__((ext_vector_type(8)))  _Float16 v8h;
typedef __attribute__((ext_vector_type(8)))  float    v8f;

#define DIMX   512
#define NHEADS 16
#define KD     32
#define VD     128
#define HQKV   192          // per-head qkv output cols (32+32+128)
#define NTOK   196
#define NP     224          // padded token count (14 tiles of 16)
#define NBATCH 256
#define NOFF   196
#define DH     2048
#define ATT_SCALE 0.17677669529663687f   // 32^-0.5

// ---------------- WMMA helpers (contiguous k = 16*(lane/16) + element in A and B) ----------------

__device__ __forceinline__ v8f wmma_f16(v16h a, v16h b, v8f c) {
  return __builtin_amdgcn_wmma_f32_16x16x32_f16(false, a, false, b, (short)0, c,
                                                false, false);
}

// A fragment (16x32 f16), row-major source, NO bounds check (buffers padded):
// lane<16 -> rows 0..15, k offsets {0..15}; lane>=16 -> k offsets {16..31}.
__device__ __forceinline__ v16h ld_a(const _Float16* base, int lda,
                                     int row0, int k0, int lane) {
  v16h a;
  int r    = row0 + (lane & 15);
  int koff = k0 + ((lane >> 4) << 4);
  v8h lo = *(const v8h*)(base + r * lda + koff);
  v8h hi = *(const v8h*)(base + r * lda + koff + 8);
#pragma unroll
  for (int i = 0; i < 8; ++i) { a[i] = lo[i]; a[i + 8] = hi[i]; }
  return a;
}

// B fragment of W^T where W is row-major [n][k] (reads are k-contiguous):
// lane holds col n0+(lane&15); elements j -> k = k0 + (lane<16?0:16) + j.
__device__ __forceinline__ v16h ld_bt(const _Float16* W, int ldw,
                                      int n0, int k0, int lane) {
  int c  = n0 + (lane & 15);
  int kb = k0 + ((lane >> 4) << 4);
  return *(const v16h*)(W + c * ldw + kb);
}

__device__ __forceinline__ v8f vzero8() {
  v8f z = {0.f, 0.f, 0.f, 0.f, 0.f, 0.f, 0.f, 0.f};
  return z;
}

// Async global->LDS copy of 16 bytes per lane (CDNA5, tracked by ASYNCcnt).
// lds_off: byte offset in LDS (addr[31:0] of the flat shared pointer);
// gbase/goff: SGPR 64-bit base + per-lane 32-bit byte offset (GVS mode).
__device__ __forceinline__ void async_copy_b128(unsigned int lds_off,
                                                unsigned long long gbase,
                                                int goff) {
  asm volatile("global_load_async_to_lds_b128 %0, %1, %2 offset:0"
               :: "v"(lds_off), "v"(goff), "s"(gbase)
               : "memory");
}

__device__ __forceinline__ void wait_async_zero() {
  asm volatile("s_wait_asynccnt 0x0" ::: "memory");
}

// ---------------- Kernel 1: LayerNorm -> f16 (row-padded to NP per batch) ----

__global__ __launch_bounds__(256)
void ln_kernel(const float* __restrict__ x, const float* __restrict__ g,
               const float* __restrict__ bb, _Float16* __restrict__ xn) {
  int wave = threadIdx.x >> 5, lane = threadIdx.x & 31;
  int p = blockIdx.x * 8 + wave;          // padded row index
  if (p >= NBATCH * NP) return;
  int b = p / NP, i = p - b * NP;
  _Float16* xo = xn + (long)p * DIMX;
  if (i >= NTOK) {                        // zero pad rows once
#pragma unroll
    for (int j = 0; j < 16; ++j) xo[lane + 32 * j] = (_Float16)0.f;
    return;
  }
  const float* xr = x + (long)(b * NTOK + i) * DIMX;
  float v[16];
  float s = 0.f;
#pragma unroll
  for (int j = 0; j < 16; ++j) { v[j] = xr[lane + 32 * j]; s += v[j]; }
#pragma unroll
  for (int m = 16; m >= 1; m >>= 1) s += __shfl_xor(s, m, 32);
  float mu = s * (1.f / 512.f);
  float q = 0.f;
#pragma unroll
  for (int j = 0; j < 16; ++j) { float d = v[j] - mu; q += d * d; }
#pragma unroll
  for (int m = 16; m >= 1; m >>= 1) q += __shfl_xor(q, m, 32);
  float inv = rsqrtf(q * (1.f / 512.f) + 1e-5f);
#pragma unroll
  for (int j = 0; j < 16; ++j) {
    int c = lane + 32 * j;
    xo[c] = (_Float16)((v[j] - mu) * inv * g[c] + bb[c]);
  }
}

// ---------------- Kernel 2: f32 -> f16 weight conversion ----------------

__global__ __launch_bounds__(256)
void cvt_kernel(const float* __restrict__ s, _Float16* __restrict__ d, long n) {
  long i = (long)blockIdx.x * 256 + threadIdx.x;
  long stride = (long)gridDim.x * 256;
  for (; i < n; i += stride) d[i] = (_Float16)s[i];
}

// ---------------- Kernel 3: fused per-(b,h) attention ----------------
// LDS: S (NP*NP f32, doubles as weight stage in Phase 1) |
//      q (NP*32 f16) | k (NP*32 f16) | vT (128*NP f16) | bias (NOFF f32)

__global__ __launch_bounds__(256, 1)
void attn_kernel(const _Float16* __restrict__ xn, const _Float16* __restrict__ wqkv,
                 const float* __restrict__ qkv_b, const float* __restrict__ att_bias,
                 const int* __restrict__ bias_idx, _Float16* __restrict__ outh) {
  extern __shared__ char smem[];
  float*    Ssm = (float*)smem;
  _Float16* qsm = (_Float16*)(smem + NP * NP * 4);
  _Float16* ksm = qsm + NP * 32;
  _Float16* vT  = ksm + NP * 32;                      // [128][NP], transposed v
  float*    bsm = (float*)(vT + 128 * NP);            // bias row for head h

  int b = blockIdx.x >> 4;
  int h = blockIdx.x & 15;
  int tid = threadIdx.x, wave = tid >> 5, lane = tid & 31;

  const _Float16* xb = xn + b * NP * DIMX;            // padded, no bounds needed
  const _Float16* wh = wqkv + h * HQKV * DIMX;

  // Phase 0: DMA the whole per-head weight block (192x512 f16 = 196,608 B)
  // into the (currently idle) S region with async global->LDS transfers.
  {
    unsigned int sbase = (unsigned int)(size_t)(void*)smem;  // LDS byte offset
    unsigned long long wsrc = (unsigned long long)(size_t)(const void*)wh;
#pragma unroll 1
    for (int it = 0; it < (HQKV * DIMX * 2) / 4096; ++it) {  // 48 iters
      int off = it * 4096 + tid * 16;
      async_copy_b128(sbase + off, wsrc, off);
    }
  }
  // Independent small inits while the DMA is in flight:
  for (int i = tid; i < NOFF; i += 256) bsm[i] = att_bias[h * NOFF + i];
  for (int i = tid; i < 128 * (NP - NTOK); i += 256) {
    int d = i / (NP - NTOK), m = NTOK + i % (NP - NTOK);
    vT[d * NP + m] = (_Float16)0.f;                   // zero vT pad (avoid NaN*0)
  }
  wait_async_zero();
  __syncthreads();

  // Phase 1: [q|k|v] = xn @ wh^T with B fragments from LDS. Wave owns an
  // M-tile, 12 N-tile accumulators; A loaded once per k-step, 12 WMMAs on it.
  const _Float16* wls = (const _Float16*)smem;        // staged weights
  for (int mt = wave; mt < 14; mt += 8) {
    v8f acc[12];
#pragma unroll
    for (int nt = 0; nt < 12; ++nt) acc[nt] = vzero8();
#pragma unroll 1
    for (int k = 0; k < DIMX; k += 32) {
      v16h a = ld_a(xb, DIMX, mt * 16, k, lane);
#pragma unroll
      for (int nt = 0; nt < 12; ++nt) {
        v16h w = ld_bt(wls, DIMX, nt * 16, k, lane);
        acc[nt] = wmma_f16(a, w, acc[nt]);
      }
    }
    int rbase = mt * 16 + ((lane >> 4) << 3);
#pragma unroll
    for (int nt = 0; nt < 12; ++nt) {
      int col  = nt * 16 + (lane & 15);
      float bv = qkv_b[h * HQKV + col];
#pragma unroll
      for (int r = 0; r < 8; ++r) {
        int n = rbase + r;
        if (n < NTOK) {
          _Float16 val = (_Float16)(acc[nt][r] + bv);
          if (col < 32)      qsm[n * 32 + col] = val;
          else if (col < 64) ksm[n * 32 + (col - 32)] = val;
          else               vT[(col - 64) * NP + n] = val;
        }
      }
    }
  }
  __syncthreads();

  // Phase 2: S = q @ k^T * scale + bias  (A reused across the 14 N-tiles).
  // Overwrites the weight stage -- safe after the barrier above.
  for (int mt = wave; mt < 14; mt += 8) {
    v16h a = ld_a(qsm, 32, mt * 16, 0, lane);
    int rbase = mt * 16 + ((lane >> 4) << 3);
#pragma unroll 1
    for (int nt = 0; nt < 14; ++nt) {
      v16h kk = ld_bt(ksm, 32, nt * 16, 0, lane);
      v8f acc = vzero8();
      acc = wmma_f16(a, kk, acc);
      int mcol = nt * 16 + (lane & 15);
#pragma unroll
      for (int r = 0; r < 8; ++r) {
        int n = rbase + r;
        float sv = acc[r] * ATT_SCALE;
        if (n < NTOK && mcol < NTOK) sv += bsm[bias_idx[n * NTOK + mcol]];
        Ssm[n * NP + mcol] = sv;
      }
    }
  }
  __syncthreads();

  // Phase 3: softmax per row; pack f16 probabilities in place (pitch 2*NP f16).
  // volatile keeps the narrowing in-place writes ordered vs. the f32 reads.
  if (tid < NP) {
    volatile _Float16* prow = (volatile _Float16*)(Ssm + tid * NP);
    if (tid < NTOK) {
      volatile float* srow = (volatile float*)(Ssm + tid * NP);
      float mx = -1e30f;
      for (int m = 0; m < NTOK; ++m) mx = fmaxf(mx, srow[m]);
      float sum = 0.f;
      for (int m = 0; m < NTOK; ++m) sum += __expf(srow[m] - mx);
      float inv = 1.f / sum;
      for (int m = 0; m < NTOK; ++m) {
        float pp = __expf(srow[m] - mx) * inv;
        prow[m] = (_Float16)pp;
      }
      for (int m = NTOK; m < NP; ++m) prow[m] = (_Float16)0.f;
    } else {
      for (int m = 0; m < NP; ++m) prow[m] = (_Float16)0.f;
    }
  }
  __syncthreads();

  // Phase 4: O = P @ v. 8 N-tile accumulators per M-tile; A loaded once per k.
  // Pad rows of P are all-zero, so pad-row outputs are exact zeros -> outh is
  // written unconditionally into its padded (NP-row) layout.
  const _Float16* Ph = (const _Float16*)Ssm;  // pitch 2*NP f16
  for (int mt = wave; mt < 14; mt += 8) {
    v8f acc[8];
#pragma unroll
    for (int nt = 0; nt < 8; ++nt) acc[nt] = vzero8();
#pragma unroll 1
    for (int k = 0; k < NP; k += 32) {
      v16h a = ld_a(Ph, 2 * NP, mt * 16, k, lane);
#pragma unroll
      for (int nt = 0; nt < 8; ++nt) {
        v16h v = ld_bt(vT, NP, nt * 16, k, lane);   // B[k=m][n=d] = vT[d][m]
        acc[nt] = wmma_f16(a, v, acc[nt]);
      }
    }
    int rbase = mt * 16 + ((lane >> 4) << 3);
#pragma unroll
    for (int nt = 0; nt < 8; ++nt) {
      int dcol = nt * 16 + (lane & 15);
#pragma unroll
      for (int r = 0; r < 8; ++r) {
        int n = rbase + r;
        outh[(b * NP + n) * DH + h * VD + dcol] = (_Float16)acc[nt][r];
      }
    }
  }
}

// ---------------- Kernel 4: output projection ----------------

__global__ __launch_bounds__(256)
void proj_kernel(const _Float16* __restrict__ outh, const _Float16* __restrict__ wp,
                 const float* __restrict__ pb, float* __restrict__ out) {
  int b = blockIdx.x;
  int tid = threadIdx.x, wave = tid >> 5, lane = tid & 31;
  const _Float16* A = outh + b * NP * DH;
  // 14 M-tiles x 4 N-chunks (8 tiles each) = 56 units, 7 per wave
  for (int t = wave; t < 14 * 4; t += 8) {
    int mt = t >> 2, nc = t & 3;
    v8f acc[8];
#pragma unroll
    for (int j = 0; j < 8; ++j) acc[j] = vzero8();
#pragma unroll 1
    for (int k = 0; k < DH; k += 32) {
      v16h a = ld_a(A, DH, mt * 16, k, lane);
#pragma unroll
      for (int j = 0; j < 8; ++j) {
        v16h w = ld_bt(wp, DH, (nc * 8 + j) * 16, k, lane);
        acc[j] = wmma_f16(a, w, acc[j]);
      }
    }
    int rbase = mt * 16 + ((lane >> 4) << 3);
#pragma unroll
    for (int j = 0; j < 8; ++j) {
      int col  = (nc * 8 + j) * 16 + (lane & 15);
      float bv = pb[col];
#pragma unroll
      for (int r = 0; r < 8; ++r) {
        int n = rbase + r;
        if (n < NTOK)
          out[((long)(b * NTOK + n)) * DIMX + col] = acc[j][r] + bv;
      }
    }
  }
}

// ---------------- launch ----------------

extern "C" void kernel_launch(void* const* d_in, const int* in_sizes, int n_in,
                              void* d_out, int out_size, void* d_ws, size_t ws_size,
                              hipStream_t stream) {
  const float* x        = (const float*)d_in[0];
  const float* ln_g     = (const float*)d_in[1];
  const float* ln_b     = (const float*)d_in[2];
  const float* qkv_w    = (const float*)d_in[3];
  const float* qkv_b    = (const float*)d_in[4];
  const float* proj_w   = (const float*)d_in[5];
  const float* proj_b   = (const float*)d_in[6];
  const float* att_bias = (const float*)d_in[7];
  const int*   bias_idx = (const int*)d_in[8];
  float* out = (float*)d_out;

  char* ws = (char*)d_ws;
  size_t off = 0;
  auto take = [&](size_t bytes) {
    char* p = ws + off;
    off = (off + bytes + 255) & ~(size_t)255;
    return p;
  };
  _Float16* xn   = (_Float16*)take((size_t)NBATCH * NP * DIMX * 2);  // padded
  _Float16* wq16 = (_Float16*)take((size_t)3072 * 512 * 2);
  _Float16* wp16 = (_Float16*)take((size_t)512 * 2048 * 2);
  _Float16* outh = (_Float16*)take((size_t)NBATCH * NP * DH * 2);    // padded
  (void)ws_size;

  int nprows = NBATCH * NP;
  ln_kernel<<<nprows / 8, 256, 0, stream>>>(x, ln_g, ln_b, xn);
  cvt_kernel<<<1536, 256, 0, stream>>>(qkv_w, wq16, (long)3072 * 512);
  cvt_kernel<<<1024, 256, 0, stream>>>(proj_w, wp16, (long)512 * 2048);

  size_t smem = (size_t)NP * NP * 4 + 2 * (size_t)NP * 32 * 2 +
                (size_t)128 * NP * 2 + (size_t)NOFF * 4;  // 287,504 B < 320 KB
  (void)hipFuncSetAttribute((const void*)attn_kernel,
                            hipFuncAttributeMaxDynamicSharedMemorySize, (int)smem);
  attn_kernel<<<NBATCH * NHEADS, 256, smem, stream>>>(xn, wq16, qkv_b, att_bias,
                                                      bias_idx, outh);
  proj_kernel<<<NBATCH, 256, 0, stream>>>(outh, wp16, proj_b, out);
  (void)out_size; (void)n_in; (void)in_sizes;
}
